// MultiHeadAttention_8040178778294
// MI455X (gfx1250) — matrix operand order repaired and checked by reading, not compiled, not run on hardware
//
#include <hip/hip_runtime.h>
#include <hip/hip_bf16.h>
#include <float.h>

typedef __attribute__((ext_vector_type(16))) __bf16 v16bf;
typedef __attribute__((ext_vector_type(8)))  float  v8f;
typedef unsigned short u16;
typedef unsigned int   u32;

// Problem constants (B=4, T=1024, D=512, H=8, dk=64)
#define TT  1024
#define DD  512
#define DK  64
#define NH  8
#define NB  4

static __device__ __forceinline__ u16 f2bf(float f) {
    u32 u = __float_as_uint(f);
    u32 r = u + 0x7FFFu + ((u >> 16) & 1u);   // round-to-nearest-even
    return (u16)(r >> 16);
}

union Frag {
    v16bf v;
    uint4 q[2];
    u32   u[8];
    u16   s[16];
};

static __device__ __forceinline__ v8f wmma_bf16(v16bf a, v16bf b, v8f c) {
    return __builtin_amdgcn_wmma_f32_16x16x32_bf16(false, a, false, b, (short)0, c, false, false);
}

// A fragment: row-major M x K (K contiguous), 16x32 bf16 tile at (row0, k0).
static __device__ __forceinline__ v16bf load_a_frag(const u16* A, long lda, int row0, int k0, int lane) {
    int half = lane >> 4;
    const u16* p = A + (long)(row0 + (lane & 15)) * lda + k0;
    Frag f;
    f.q[0] = *(const uint4*)(p + half * 16);
    f.q[1] = *(const uint4*)(p + half * 16 + 8);
    return f.v;
}

// B fragment: B[k][n] = W[n*ldb + k]  (N-major, K contiguous), 32x16 tile at (k0, n0).
static __device__ __forceinline__ v16bf load_b_frag_kt(const u16* W, long ldb, int k0, int n0, int lane) {
    int half = lane >> 4;
    const u16* p = W + (long)(n0 + (lane & 15)) * ldb + k0 + half * 16;
    Frag f;
    f.q[0] = *(const uint4*)(p);
    f.q[1] = *(const uint4*)(p + 8);
    return f.v;
}

// Same KT layout from an f32 source, converting to bf16 on the fly.
static __device__ __forceinline__ v16bf load_b_frag_kt_f32(const float* W, long ldb, int k0, int n0, int lane) {
    int half = lane >> 4;
    const float* p = W + (long)(n0 + (lane & 15)) * ldb + k0 + half * 16;
    const float4* p4 = (const float4*)p;
    float4 a = p4[0], b = p4[1], c = p4[2], d = p4[3];
    Frag f;
    f.u[0] = (u32)f2bf(a.x) | ((u32)f2bf(a.y) << 16);
    f.u[1] = (u32)f2bf(a.z) | ((u32)f2bf(a.w) << 16);
    f.u[2] = (u32)f2bf(b.x) | ((u32)f2bf(b.y) << 16);
    f.u[3] = (u32)f2bf(b.z) | ((u32)f2bf(b.w) << 16);
    f.u[4] = (u32)f2bf(c.x) | ((u32)f2bf(c.y) << 16);
    f.u[5] = (u32)f2bf(c.z) | ((u32)f2bf(c.w) << 16);
    f.u[6] = (u32)f2bf(d.x) | ((u32)f2bf(d.y) << 16);
    f.u[7] = (u32)f2bf(d.z) | ((u32)f2bf(d.w) << 16);
    return f.v;
}

// ---------------- Kernel 1: LayerNorm -> xn (bf16) ----------------
__global__ void ln_kernel(const float* __restrict__ x, const float* __restrict__ g,
                          const float* __restrict__ bta, u16* __restrict__ xn) {
    __shared__ float red[256];
    int row = blockIdx.x, tid = threadIdx.x;
    const float* xr = x + (long)row * DD;
    float x0 = xr[tid], x1 = xr[tid + 256];
    red[tid] = x0 + x1;
    __syncthreads();
    for (int st = 128; st > 0; st >>= 1) { if (tid < st) red[tid] += red[tid + st]; __syncthreads(); }
    float mu = red[0] * (1.0f / DD);
    __syncthreads();
    red[tid] = x0 * x0 + x1 * x1;
    __syncthreads();
    for (int st = 128; st > 0; st >>= 1) { if (tid < st) red[tid] += red[tid + st]; __syncthreads(); }
    float var = red[0] * (1.0f / DD) - mu * mu;
    float rs = rsqrtf(var + 1e-5f);
    u16* xo = xn + (long)row * DD;
    xo[tid]       = f2bf((x0 - mu) * rs * g[tid]       + bta[tid]);
    xo[tid + 256] = f2bf((x1 - mu) * rs * g[tid + 256] + bta[tid + 256]);
}

// ---------------- Kernel 2: convert Wq|Wk|Wv|Wo to bf16 ----------------
__global__ void wconv_kernel(const float* __restrict__ w0, const float* __restrict__ w1,
                             const float* __restrict__ w2, const float* __restrict__ w3,
                             u16* __restrict__ dst) {
    int i = blockIdx.x * blockDim.x + threadIdx.x;
    if (i >= 4 * DD * DD) return;
    int sel = i >> 18, off = i & (DD * DD - 1);
    const float* s = sel == 0 ? w0 : sel == 1 ? w1 : sel == 2 ? w2 : w3;
    dst[i] = f2bf(s[off]);
}

// ---------------- Kernel 3: QKV projection GEMMs (32x64 tile per wave) -------
// y = xn @ W^T + b ; q,k laid out (B,H,T,dk); v laid out transposed (B,H,dk,T)
static __device__ __forceinline__ void qkv_store(v8f acc, int m0, int n0, int lane, int sel,
                                                 const float* __restrict__ bias,
                                                 u16* __restrict__ qo, u16* __restrict__ ko,
                                                 u16* __restrict__ vo) {
    int half = lane >> 4, col = lane & 15;
    for (int r = 0; r < 8; ++r) {
        int m = m0 + r + half * 8, n = n0 + col;
        float val = acc[r] + bias[n];
        int bb = m >> 10, t = m & (TT - 1);
        int h = n >> 6, d = n & (DK - 1);
        u16 v16 = f2bf(val);
        if (sel == 0)      qo[(((long)(bb * NH + h)) * TT + t) * DK + d] = v16;
        else if (sel == 1) ko[(((long)(bb * NH + h)) * TT + t) * DK + d] = v16;
        else               vo[(((long)(bb * NH + h)) * DK + d) * TT + t] = v16;
    }
}

__global__ void qkv_gemm(const u16* __restrict__ xn, const u16* __restrict__ wbf,
                         const float* __restrict__ bq, const float* __restrict__ bk,
                         const float* __restrict__ bv,
                         u16* __restrict__ qo, u16* __restrict__ ko, u16* __restrict__ vo) {
    int lane = threadIdx.x & 31, w = threadIdx.x >> 5;
    int id = blockIdx.x * 4 + w;                 // [0, 1024)
    int mtile = id >> 3, ngrp = id & 7;          // 128 x 8
    int sel = blockIdx.y;
    const u16* W = wbf + (long)sel * (DD * DD);
    const float* bias = sel == 0 ? bq : sel == 1 ? bk : bv;
    int m0 = mtile * 32, n0 = ngrp * 64;
    v8f a00 = {}, a01 = {}, a02 = {}, a03 = {};
    v8f a10 = {}, a11 = {}, a12 = {}, a13 = {};
    for (int k0 = 0; k0 < DD; k0 += 32) {
        v16bf fa0 = load_a_frag(xn, DD, m0,      k0, lane);
        v16bf fa1 = load_a_frag(xn, DD, m0 + 16, k0, lane);
        v16bf fb0 = load_b_frag_kt(W, DD, k0, n0,      lane);
        v16bf fb1 = load_b_frag_kt(W, DD, k0, n0 + 16, lane);
        v16bf fb2 = load_b_frag_kt(W, DD, k0, n0 + 32, lane);
        v16bf fb3 = load_b_frag_kt(W, DD, k0, n0 + 48, lane);
        a00 = wmma_bf16(fa0, fb0, a00); a01 = wmma_bf16(fa0, fb1, a01);
        a02 = wmma_bf16(fa0, fb2, a02); a03 = wmma_bf16(fa0, fb3, a03);
        a10 = wmma_bf16(fa1, fb0, a10); a11 = wmma_bf16(fa1, fb1, a11);
        a12 = wmma_bf16(fa1, fb2, a12); a13 = wmma_bf16(fa1, fb3, a13);
    }
    qkv_store(a00, m0,      n0,      lane, sel, bias, qo, ko, vo);
    qkv_store(a01, m0,      n0 + 16, lane, sel, bias, qo, ko, vo);
    qkv_store(a02, m0,      n0 + 32, lane, sel, bias, qo, ko, vo);
    qkv_store(a03, m0,      n0 + 48, lane, sel, bias, qo, ko, vo);
    qkv_store(a10, m0 + 16, n0,      lane, sel, bias, qo, ko, vo);
    qkv_store(a11, m0 + 16, n0 + 16, lane, sel, bias, qo, ko, vo);
    qkv_store(a12, m0 + 16, n0 + 32, lane, sel, bias, qo, ko, vo);
    qkv_store(a13, m0 + 16, n0 + 48, lane, sel, bias, qo, ko, vo);
}

// ---------------- Kernel 4: scores = Q @ K^T (per b,h), f32; 32x64 per wave ---
static __device__ __forceinline__ void sc_store(v8f acc, int i0, int j0, int lane,
                                                float* __restrict__ srow) {
    int half = lane >> 4, col = lane & 15;
    for (int r = 0; r < 8; ++r) {
        int i = i0 + r + half * 8, j = j0 + col;
        srow[(long)i * TT + j] = acc[r];
    }
}

__global__ void scores_gemm(const u16* __restrict__ q, const u16* __restrict__ k,
                            float* __restrict__ s) {
    int lane = threadIdx.x & 31, w = threadIdx.x >> 5;
    int bh = blockIdx.y;
    int id = blockIdx.x * 4 + w;                 // [0, 512)
    int itile = id >> 4, ngrp = id & 15;         // 32 x 16
    int i0 = itile * 32, j0 = ngrp * 64;
    const u16* A  = q + (long)bh * (TT * DK);
    const u16* Bm = k + (long)bh * (TT * DK);
    v8f a00 = {}, a01 = {}, a02 = {}, a03 = {};
    v8f a10 = {}, a11 = {}, a12 = {}, a13 = {};
    for (int k0 = 0; k0 < DK; k0 += 32) {
        v16bf fa0 = load_a_frag(A, DK, i0,      k0, lane);
        v16bf fa1 = load_a_frag(A, DK, i0 + 16, k0, lane);
        v16bf fb0 = load_b_frag_kt(Bm, DK, k0, j0,      lane);
        v16bf fb1 = load_b_frag_kt(Bm, DK, k0, j0 + 16, lane);
        v16bf fb2 = load_b_frag_kt(Bm, DK, k0, j0 + 32, lane);
        v16bf fb3 = load_b_frag_kt(Bm, DK, k0, j0 + 48, lane);
        a00 = wmma_bf16(fa0, fb0, a00); a01 = wmma_bf16(fa0, fb1, a01);
        a02 = wmma_bf16(fa0, fb2, a02); a03 = wmma_bf16(fa0, fb3, a03);
        a10 = wmma_bf16(fa1, fb0, a10); a11 = wmma_bf16(fa1, fb1, a11);
        a12 = wmma_bf16(fa1, fb2, a12); a13 = wmma_bf16(fa1, fb3, a13);
    }
    float* srow = s + (long)bh * TT * TT;
    sc_store(a00, i0,      j0,      lane, srow);
    sc_store(a01, i0,      j0 + 16, lane, srow);
    sc_store(a02, i0,      j0 + 32, lane, srow);
    sc_store(a03, i0,      j0 + 48, lane, srow);
    sc_store(a10, i0 + 16, j0,      lane, srow);
    sc_store(a11, i0 + 16, j0 + 16, lane, srow);
    sc_store(a12, i0 + 16, j0 + 32, lane, srow);
    sc_store(a13, i0 + 16, j0 + 48, lane, srow);
}

// ---------------- Kernel 5: scores += Q_i @ pos_k[i]^T ----------------
// For fixed i: A = Q rows over (b,h) [32 x 64], B = pos_k[i] [64 x T]
__global__ void rel_gemm(const u16* __restrict__ q, const float* __restrict__ pos_k,
                         float* __restrict__ s) {
    int lane = threadIdx.x & 31, w = threadIdx.x >> 5;
    int i = blockIdx.x;
    int jtile = blockIdx.y * 4 + w;
    int j0 = jtile * 16;
    const u16* A = q + (long)i * DK;                       // row stride T*DK over bh
    const float* Bp = pos_k + (long)i * (TT * DK);
    v16bf b0 = load_b_frag_kt_f32(Bp, DK, 0, j0, lane);
    v16bf b1 = load_b_frag_kt_f32(Bp, DK, 32, j0, lane);
    v16bf a00 = load_a_frag(A, (long)TT * DK, 0,  0,  lane);
    v16bf a01 = load_a_frag(A, (long)TT * DK, 0,  32, lane);
    v16bf a10 = load_a_frag(A, (long)TT * DK, 16, 0,  lane);
    v16bf a11 = load_a_frag(A, (long)TT * DK, 16, 32, lane);
    v8f acc0 = {}, acc1 = {};
    acc0 = wmma_bf16(a00, b0, acc0);
    acc0 = wmma_bf16(a01, b1, acc0);
    acc1 = wmma_bf16(a10, b0, acc1);
    acc1 = wmma_bf16(a11, b1, acc1);
    int half = lane >> 4, col = lane & 15;
    for (int mt = 0; mt < 2; ++mt) {
        v8f acc = mt == 0 ? acc0 : acc1;
        for (int r = 0; r < 8; ++r) {
            int bh = mt * 16 + r + half * 8;
            int j = j0 + col;
            long idx = (long)bh * TT * TT + (long)i * TT + j;
            s[idx] += acc[r];
        }
    }
}

// ---------------- Kernel 6: masked softmax -> bf16 attn (aliased over scores) ----
__global__ void softmax_kernel(const int* __restrict__ mask, float* __restrict__ s) {
    __shared__ float red[256];
    int R = blockIdx.x;                   // bh*T + i
    int bh = R >> 10, i = R & (TT - 1);
    int b = bh >> 3;
    int tid = threadIdx.x;
    const int* mrow = mask + (long)b * TT * TT + (long)i * TT;
    long base = (long)R * TT;
    const float inv = 0.125f;             // 1/sqrt(dk)
    float v[4]; int mk[4];
    float lmax = -FLT_MAX;
    for (int c = 0; c < 4; ++c) {
        int j = tid + c * 256;
        mk[c] = mrow[j];
        float raw = s[base + j];
        v[c] = (mk[c] != 0) ? raw * inv : -FLT_MAX;
        lmax = fmaxf(lmax, v[c]);
    }
    red[tid] = lmax;
    __syncthreads();
    for (int st = 128; st > 0; st >>= 1) { if (tid < st) red[tid] = fmaxf(red[tid], red[tid + st]); __syncthreads(); }
    float rmax = red[0];
    __syncthreads();
    float e[4]; float lsum = 0.f;
    for (int c = 0; c < 4; ++c) {
        e[c] = (mk[c] != 0) ? __expf(v[c] - rmax) : 0.f;
        lsum += e[c];
    }
    red[tid] = lsum;
    __syncthreads();
    for (int st = 128; st > 0; st >>= 1) { if (tid < st) red[tid] += red[tid + st]; __syncthreads(); }
    float rsum = red[0];
    float rinv = rsum > 0.f ? 1.f / rsum : 0.f;
    // attn row R lives in the first half (2KB) of the 4KB f32 scores row R
    u16* attn = (u16*)s;
    long abase = (long)R * (2 * TT);
    for (int c = 0; c < 4; ++c) attn[abase + tid + c * 256] = f2bf(e[c] * rinv);
}

// ---------------- Kernel 7: ctx = attn @ V (per b,h) -> bf16 (B,T,D) ----------
// Each wave computes a 32 x 64 tile (all of dk) for 32 query rows.
static __device__ __forceinline__ void av_store(v8f acc, int t0, int d0, int lane,
                                                int b, int h, u16* __restrict__ ctx) {
    int half = lane >> 4, col = lane & 15;
    for (int r = 0; r < 8; ++r) {
        int t = t0 + r + half * 8, d = d0 + col;
        ctx[((long)(b * TT + t)) * DD + h * DK + d] = f2bf(acc[r]);
    }
}

__global__ void av_gemm(const u16* __restrict__ attn, const u16* __restrict__ vt,
                        u16* __restrict__ ctx) {
    int lane = threadIdx.x & 31, w = threadIdx.x >> 5;
    int bh = blockIdx.y;
    int mtile = blockIdx.x * 4 + w;              // [0, 32)
    int m0 = mtile * 32;
    const u16* A  = attn + (long)bh * TT * (2 * TT);     // lda = 2*T (aliased rows)
    const u16* Bv = vt + (long)bh * DK * TT;             // B[k=j][n=d] = vt[d*T + j]
    v8f a00 = {}, a01 = {}, a02 = {}, a03 = {};
    v8f a10 = {}, a11 = {}, a12 = {}, a13 = {};
    for (int k0 = 0; k0 < TT; k0 += 32) {
        v16bf fa0 = load_a_frag(A, 2 * TT, m0,      k0, lane);
        v16bf fa1 = load_a_frag(A, 2 * TT, m0 + 16, k0, lane);
        v16bf fb0 = load_b_frag_kt(Bv, TT, k0, 0,  lane);
        v16bf fb1 = load_b_frag_kt(Bv, TT, k0, 16, lane);
        v16bf fb2 = load_b_frag_kt(Bv, TT, k0, 32, lane);
        v16bf fb3 = load_b_frag_kt(Bv, TT, k0, 48, lane);
        a00 = wmma_bf16(fa0, fb0, a00); a01 = wmma_bf16(fa0, fb1, a01);
        a02 = wmma_bf16(fa0, fb2, a02); a03 = wmma_bf16(fa0, fb3, a03);
        a10 = wmma_bf16(fa1, fb0, a10); a11 = wmma_bf16(fa1, fb1, a11);
        a12 = wmma_bf16(fa1, fb2, a12); a13 = wmma_bf16(fa1, fb3, a13);
    }
    int b = bh >> 3, h = bh & 7;
    av_store(a00, m0,      0,  lane, b, h, ctx);
    av_store(a01, m0,      16, lane, b, h, ctx);
    av_store(a02, m0,      32, lane, b, h, ctx);
    av_store(a03, m0,      48, lane, b, h, ctx);
    av_store(a10, m0 + 16, 0,  lane, b, h, ctx);
    av_store(a11, m0 + 16, 16, lane, b, h, ctx);
    av_store(a12, m0 + 16, 32, lane, b, h, ctx);
    av_store(a13, m0 + 16, 48, lane, b, h, ctx);
}

// ---------------- Kernel 8: out = (ctx @ Wo^T + bo) * layer_scale ------------
static __device__ __forceinline__ void out_store(v8f acc, int m0, int n0, int lane,
                                                 const float* __restrict__ bo,
                                                 const float* __restrict__ ls,
                                                 float* __restrict__ out) {
    int half = lane >> 4, col = lane & 15;
    for (int r = 0; r < 8; ++r) {
        int m = m0 + r + half * 8, n = n0 + col;
        out[(long)m * DD + n] = (acc[r] + bo[n]) * ls[n];
    }
}

__global__ void out_gemm(const u16* __restrict__ ctx, const u16* __restrict__ wo,
                         const float* __restrict__ bo, const float* __restrict__ ls,
                         float* __restrict__ out) {
    int lane = threadIdx.x & 31, w = threadIdx.x >> 5;
    int id = blockIdx.x * 4 + w;                 // [0, 1024)
    int mtile = id >> 3, ngrp = id & 7;          // 128 x 8
    int m0 = mtile * 32, n0 = ngrp * 64;
    v8f a00 = {}, a01 = {}, a02 = {}, a03 = {};
    v8f a10 = {}, a11 = {}, a12 = {}, a13 = {};
    for (int k0 = 0; k0 < DD; k0 += 32) {
        v16bf fa0 = load_a_frag(ctx, DD, m0,      k0, lane);
        v16bf fa1 = load_a_frag(ctx, DD, m0 + 16, k0, lane);
        v16bf fb0 = load_b_frag_kt(wo, DD, k0, n0,      lane);
        v16bf fb1 = load_b_frag_kt(wo, DD, k0, n0 + 16, lane);
        v16bf fb2 = load_b_frag_kt(wo, DD, k0, n0 + 32, lane);
        v16bf fb3 = load_b_frag_kt(wo, DD, k0, n0 + 48, lane);
        a00 = wmma_bf16(fa0, fb0, a00); a01 = wmma_bf16(fa0, fb1, a01);
        a02 = wmma_bf16(fa0, fb2, a02); a03 = wmma_bf16(fa0, fb3, a03);
        a10 = wmma_bf16(fa1, fb0, a10); a11 = wmma_bf16(fa1, fb1, a11);
        a12 = wmma_bf16(fa1, fb2, a12); a13 = wmma_bf16(fa1, fb3, a13);
    }
    out_store(a00, m0,      n0,      lane, bo, ls, out);
    out_store(a01, m0,      n0 + 16, lane, bo, ls, out);
    out_store(a02, m0,      n0 + 32, lane, bo, ls, out);
    out_store(a03, m0,      n0 + 48, lane, bo, ls, out);
    out_store(a10, m0 + 16, n0,      lane, bo, ls, out);
    out_store(a11, m0 + 16, n0 + 16, lane, bo, ls, out);
    out_store(a12, m0 + 16, n0 + 32, lane, bo, ls, out);
    out_store(a13, m0 + 16, n0 + 48, lane, bo, ls, out);
}

extern "C" void kernel_launch(void* const* d_in, const int* in_sizes, int n_in,
                              void* d_out, int out_size, void* d_ws, size_t ws_size,
                              hipStream_t stream) {
    const float* x     = (const float*)d_in[0];
    const float* pos_k = (const float*)d_in[1];
    const int*   mask  = (const int*)d_in[2];
    const float* ln_g  = (const float*)d_in[3];
    const float* ln_b  = (const float*)d_in[4];
    const float* Wq    = (const float*)d_in[5];
    const float* bq    = (const float*)d_in[6];
    const float* Wk    = (const float*)d_in[7];
    const float* bk    = (const float*)d_in[8];
    const float* Wv    = (const float*)d_in[9];
    const float* bv    = (const float*)d_in[10];
    const float* Wo    = (const float*)d_in[11];
    const float* bo    = (const float*)d_in[12];
    const float* ls    = (const float*)d_in[13];
    float* out = (float*)d_out;

    char* ws = (char*)d_ws;
    size_t o = 0;
    u16* xn  = (u16*)(ws + o); o += (size_t)NB * TT * DD * 2;          // 4 MiB
    u16* wbf = (u16*)(ws + o); o += (size_t)4 * DD * DD * 2;           // 2 MiB
    u16* qb  = (u16*)(ws + o); o += (size_t)NB * TT * DD * 2;          // 4 MiB
    u16* kb  = (u16*)(ws + o); o += (size_t)NB * TT * DD * 2;          // 4 MiB
    u16* vt  = (u16*)(ws + o); o += (size_t)NB * TT * DD * 2;          // 4 MiB
    float* sc = (float*)(ws + o); o += (size_t)NB * NH * TT * TT * 4;  // 128 MiB (attn aliases)
    u16* ctx = (u16*)(ws + o); o += (size_t)NB * TT * DD * 2;          // 4 MiB

    ln_kernel<<<NB * TT, 256, 0, stream>>>(x, ln_g, ln_b, xn);
    wconv_kernel<<<(4 * DD * DD) / 256, 256, 0, stream>>>(Wq, Wk, Wv, Wo, wbf);
    qkv_gemm<<<dim3(256, 3), 128, 0, stream>>>(xn, wbf, bq, bk, bv, qb, kb, vt);
    scores_gemm<<<dim3(128, NB * NH), 128, 0, stream>>>(qb, kb, sc);
    rel_gemm<<<dim3(TT, 16), 128, 0, stream>>>(qb, pos_k, sc);
    softmax_kernel<<<NB * NH * TT, 256, 0, stream>>>(mask, sc);
    av_gemm<<<dim3(8, NB * NH), 128, 0, stream>>>((const u16*)sc, vt, ctx);
    out_gemm<<<256, 128, 0, stream>>>(ctx, wbf + (size_t)3 * DD * DD, bo, ls, out);
}
